// GATResNet_68547678044310
// MI455X (gfx1250) — matrix operand order repaired and checked by reading, compiled, not run on hardware
//
#include <hip/hip_runtime.h>
#include <hip/hip_bf16.h>
#include <math.h>

// ---------------------------------------------------------------------------
// GAT ResNet for MI455X (gfx1250).
// Dense projections: f32->f16 pre-conversion into zero-padded buffers, then a
// branch-free WMMA GEMM (v_wmma_f32_16x16x32_f16) whose fragments are loaded
// directly from global memory as aligned b128 accesses (no LDS, no guards).
// Graph aggregation: coalesced loads + f32 atomics (HBM-bound part).
// ---------------------------------------------------------------------------

typedef _Float16 v16h __attribute__((ext_vector_type(16)));
typedef _Float16 v8h  __attribute__((ext_vector_type(8)));
typedef float    v8f  __attribute__((ext_vector_type(8)));

#define GAT_IN   165
#define GAT_DIM  800   // 8 heads * 100 ch for layers 1/2

#define BM 128         // rows per block (8 waves x 16)
#define BN 64          // cols per block (4 wmma subtiles per wave)
#define BK 32

// ---------------- f32 -> f16 padded row-major copy:  out[Mp][Kp] ------------
__global__ void k_cvt_pad_f16(const float* __restrict__ in, _Float16* __restrict__ outp,
                              int M, int K, int Mp, int Kp)
{
    long long n = (long long)Mp * Kp;
    long long i = (long long)blockIdx.x * blockDim.x + threadIdx.x;
    long long stride = (long long)gridDim.x * blockDim.x;
    for (; i < n; i += stride) {
        int r = (int)(i / Kp), c = (int)(i % Kp);
        float v = (r < M && c < K) ? in[(size_t)r * K + c] : 0.0f;
        outp[i] = (_Float16)v;
    }
}

// ---------- W[K][Nout] f32 -> transposed padded f16:  out[Np][Kp] -----------
__global__ void k_cvt_w_t16(const float* __restrict__ Wm, _Float16* __restrict__ outp,
                            int K, int Nout, int Np, int Kp)
{
    long long n = (long long)Np * Kp;
    long long i = (long long)blockIdx.x * blockDim.x + threadIdx.x;
    long long stride = (long long)gridDim.x * blockDim.x;
    for (; i < n; i += stride) {
        int nn = (int)(i / Kp), kk = (int)(i % Kp);
        float v = (nn < Nout && kk < K) ? Wm[(size_t)kk * Nout + nn] : 0.0f;
        outp[i] = (_Float16)v;
    }
}

// ---------------- WMMA GEMM:  C[M,Nout] = A16[Mp,Kp] * Wt16[Np,Kp]^T --------
// A16 zero-padded to Mp (mult of 128) x Kp (mult of 32); Wt16 to Np (mult of
// 64) x Kp.  All fragment loads unconditional + 16B aligned.
__global__ __launch_bounds__(256) void k_gemm_wmma(
    const _Float16* __restrict__ A16, const _Float16* __restrict__ Wt16,
    float* __restrict__ C, int M, int Kp, int Nout)
{
    const int wave = threadIdx.x >> 5;
    const int lane = threadIdx.x & 31;
    const int l15  = lane & 15;
    const int m0   = blockIdx.x * BM + wave * 16;
    const int n0   = blockIdx.y * BN;

    // A fragment (contiguous k order, same as B): lane row = m0+l15,
    // kbase = (lane>=16)?16:0; k=kb+j for j=0..15.
    const _Float16* arow = A16 + (size_t)(m0 + l15) * Kp + ((lane >= 16) ? 16 : 0);
    // B fragment (contiguous k order, same as A): lanes0-15 K=0..15, lanes16-31 K=16..31.
    const int kbB = (lane >= 16) ? 16 : 0;
    const _Float16* brow0 = Wt16 + (size_t)(n0 +  0 + l15) * Kp + kbB;
    const _Float16* brow1 = Wt16 + (size_t)(n0 + 16 + l15) * Kp + kbB;
    const _Float16* brow2 = Wt16 + (size_t)(n0 + 32 + l15) * Kp + kbB;
    const _Float16* brow3 = Wt16 + (size_t)(n0 + 48 + l15) * Kp + kbB;

    v8f acc0 = {}, acc1 = {}, acc2 = {}, acc3 = {};

    const int kTiles = Kp >> 5;
    for (int kt = 0; kt < kTiles; ++kt) {
        const int ko = kt * BK;

        v8h a_lo = *(const v8h*)(arow + ko);
        v8h a_hi = *(const v8h*)(arow + ko + 8);
        v16h af = __builtin_shufflevector(a_lo, a_hi,
            0,1,2,3,4,5,6,7,8,9,10,11,12,13,14,15);

        v8h b0l = *(const v8h*)(brow0 + ko), b0h = *(const v8h*)(brow0 + ko + 8);
        v8h b1l = *(const v8h*)(brow1 + ko), b1h = *(const v8h*)(brow1 + ko + 8);
        v8h b2l = *(const v8h*)(brow2 + ko), b2h = *(const v8h*)(brow2 + ko + 8);
        v8h b3l = *(const v8h*)(brow3 + ko), b3h = *(const v8h*)(brow3 + ko + 8);

        if (kt + 1 < kTiles) {
            __builtin_prefetch(arow + ko + BK, 0, 1);
            __builtin_prefetch(brow0 + ko + BK, 0, 1);
        }

        v16h bf0 = __builtin_shufflevector(b0l, b0h, 0,1,2,3,4,5,6,7,8,9,10,11,12,13,14,15);
        v16h bf1 = __builtin_shufflevector(b1l, b1h, 0,1,2,3,4,5,6,7,8,9,10,11,12,13,14,15);
        v16h bf2 = __builtin_shufflevector(b2l, b2h, 0,1,2,3,4,5,6,7,8,9,10,11,12,13,14,15);
        v16h bf3 = __builtin_shufflevector(b3l, b3h, 0,1,2,3,4,5,6,7,8,9,10,11,12,13,14,15);

        acc0 = __builtin_amdgcn_wmma_f32_16x16x32_f16(false, af, false, bf0, (short)0, acc0, false, false);
        acc1 = __builtin_amdgcn_wmma_f32_16x16x32_f16(false, af, false, bf1, (short)0, acc1, false, false);
        acc2 = __builtin_amdgcn_wmma_f32_16x16x32_f16(false, af, false, bf2, (short)0, acc2, false, false);
        acc3 = __builtin_amdgcn_wmma_f32_16x16x32_f16(false, af, false, bf3, (short)0, acc3, false, false);
    }

    // store: C/D layout: lane col = lane&15, vgpr r -> m = r + (lane>=16?8:0)
    const int mbase = m0 + ((lane >= 16) ? 8 : 0);
    #pragma unroll
    for (int s = 0; s < 4; ++s) {
        int gn = n0 + s * 16 + l15;
        if (gn >= Nout) continue;
        v8f acc = (s == 0) ? acc0 : (s == 1) ? acc1 : (s == 2) ? acc2 : acc3;
        #pragma unroll
        for (int r = 0; r < 8; ++r) {
            int gm = mbase + r;
            if (gm < M) C[(size_t)gm * Nout + gn] = acc[r];
        }
    }
}

// ---------------- utility fills ----------------
__global__ void k_fill(float* __restrict__ p, long long n, float v) {
    long long i = (long long)blockIdx.x * blockDim.x + threadIdx.x;
    long long stride = (long long)gridDim.x * blockDim.x;
    for (; i < n; i += stride) p[i] = v;
}

__global__ void k_init_bias(float* __restrict__ p, const float* __restrict__ b,
                            long long n, int D) {
    long long i = (long long)blockIdx.x * blockDim.x + threadIdx.x;
    long long stride = (long long)gridDim.x * blockDim.x;
    for (; i < n; i += stride) p[i] = b[i % D];
}

__global__ void k_add_bias(float* __restrict__ p, const float* __restrict__ b,
                           long long n, int D) {
    long long i = (long long)blockIdx.x * blockDim.x + threadIdx.x;
    long long stride = (long long)gridDim.x * blockDim.x;
    for (; i < n; i += stride) p[i] += b[i % D];
}

// ---------------- attention coefficients: e_src/e_dst per (node, head) ------
__global__ void k_att_coef(const float* __restrict__ h, const float* __restrict__ as,
                           const float* __restrict__ ad, float* __restrict__ es,
                           float* __restrict__ ed, int N, int heads, int ch)
{
    int node = blockIdx.x;
    int head = threadIdx.x >> 5;
    int lane = threadIdx.x & 31;
    if (node >= N || head >= heads) return;
    const float* hp = h + (size_t)node * heads * ch + (size_t)head * ch;
    const float* a1 = as + head * ch;
    const float* a2 = ad + head * ch;
    float s1 = 0.f, s2 = 0.f;
    for (int c = lane; c < ch; c += 32) {
        float v = hp[c];
        s1 += v * a1[c];
        s2 += v * a2[c];
    }
    #pragma unroll
    for (int off = 16; off > 0; off >>= 1) {
        s1 += __shfl_down(s1, off, 32);
        s2 += __shfl_down(s2, off, 32);
    }
    if (lane == 0) {
        es[(size_t)node * heads + head] = s1;
        ed[(size_t)node * heads + head] = s2;
    }
}

__device__ __forceinline__ void atomicMaxFloat(float* addr, float v) {
    if (v >= 0.0f) atomicMax((int*)addr, __float_as_int(v));
    else           atomicMin((unsigned int*)addr, __float_as_uint(v));
}

// thread per (edge, head): LeakyReLU logit + segment max
__global__ void k_edge_logits(const int* __restrict__ ei, int E, int Et,
                              const float* __restrict__ es, const float* __restrict__ ed,
                              float* __restrict__ elog, float* __restrict__ mx, int heads)
{
    long long i = (long long)blockIdx.x * blockDim.x + threadIdx.x;
    if (i >= (long long)Et * heads) return;
    int e = (int)(i / heads);
    int h = (int)(i % heads);
    int s, d;
    if (e < E) { s = ei[e]; d = ei[E + e]; } else { s = d = e - E; }
    float l = es[(size_t)s * heads + h] + ed[(size_t)d * heads + h];
    l = (l > 0.0f) ? l : 0.2f * l;
    elog[i] = l;
    atomicMaxFloat(&mx[(size_t)d * heads + h], l);
}

// thread per (edge, head): exp(l - m) + segment sum (stored over elog)
__global__ void k_edge_exp(const int* __restrict__ ei, int E, int Et,
                           float* __restrict__ elog, const float* __restrict__ mx,
                           float* __restrict__ dn, int heads)
{
    long long i = (long long)blockIdx.x * blockDim.x + threadIdx.x;
    if (i >= (long long)Et * heads) return;
    int e = (int)(i / heads);
    int h = (int)(i % heads);
    int d = (e < E) ? ei[E + e] : (e - E);
    float x = __expf(elog[i] - mx[(size_t)d * heads + h]);
    elog[i] = x;
    atomicAdd(&dn[(size_t)d * heads + h], x);
}

// wave per edge: agg[dst, :] += alpha * h[src, :]   (D = heads*ch)
__global__ void k_edge_agg(const int* __restrict__ ei, int E, int Et,
                           const float* __restrict__ ex, const float* __restrict__ dn,
                           const float* __restrict__ hp, float* __restrict__ agg,
                           int heads, int ch)
{
    int e = (int)((blockIdx.x * (long long)blockDim.x + threadIdx.x) >> 5);
    int lane = threadIdx.x & 31;
    if (e >= Et) return;
    int s, d;
    if (e < E) { s = ei[e]; d = ei[E + e]; } else { s = d = e - E; }
    const int D = heads * ch;
    for (int c = lane; c < D; c += 32) {
        int h = c / ch;
        float coef = ex[(size_t)e * heads + h] /
                     (dn[(size_t)d * heads + h] + 1e-16f);
        atomicAdd(&agg[(size_t)d * D + c], coef * hp[(size_t)s * D + c]);
    }
}

// ---------------- elementwise ELU (in place) ----------------
__global__ void k_elu(float* __restrict__ p, long long n) {
    long long i = (long long)blockIdx.x * blockDim.x + threadIdx.x;
    long long stride = (long long)gridDim.x * blockDim.x;
    for (; i < n; i += stride) {
        float v = p[i];
        p[i] = (v > 0.0f) ? v : (__expf(v) - 1.0f);
    }
}

// ---------------- wave-per-row GEMM for Nout==2 ----------------
__global__ void k_rowdot2(const float* __restrict__ A, const float* __restrict__ W,
                          float* __restrict__ out, int M, int K)
{
    int n = (int)((blockIdx.x * (long long)blockDim.x + threadIdx.x) >> 5);
    int lane = threadIdx.x & 31;
    if (n >= M) return;
    float s0 = 0.f, s1 = 0.f;
    const float* row = A + (size_t)n * K;
    for (int k = lane; k < K; k += 32) {
        float a = row[k];
        s0 += a * W[k * 2 + 0];
        s1 += a * W[k * 2 + 1];
    }
    #pragma unroll
    for (int off = 16; off > 0; off >>= 1) {
        s0 += __shfl_down(s0, off, 32);
        s1 += __shfl_down(s1, off, 32);
    }
    if (lane == 0) { out[(size_t)n * 2 + 0] = s0; out[(size_t)n * 2 + 1] = s1; }
}

// final: logits = agg3 + h2@Wres2 + x@Wskip
__global__ void k_final(const float* __restrict__ agg3, const float* __restrict__ h2,
                        const float* __restrict__ Wres2, const float* __restrict__ x,
                        const float* __restrict__ Wskip, float* __restrict__ out,
                        int N, int K2, int KX)
{
    int n = (int)((blockIdx.x * (long long)blockDim.x + threadIdx.x) >> 5);
    int lane = threadIdx.x & 31;
    if (n >= N) return;
    float s0 = 0.f, s1 = 0.f;
    const float* r2 = h2 + (size_t)n * K2;
    for (int k = lane; k < K2; k += 32) {
        float a = r2[k];
        s0 += a * Wres2[k * 2 + 0];
        s1 += a * Wres2[k * 2 + 1];
    }
    const float* rx = x + (size_t)n * KX;
    for (int k = lane; k < KX; k += 32) {
        float a = rx[k];
        s0 += a * Wskip[k * 2 + 0];
        s1 += a * Wskip[k * 2 + 1];
    }
    #pragma unroll
    for (int off = 16; off > 0; off >>= 1) {
        s0 += __shfl_down(s0, off, 32);
        s1 += __shfl_down(s1, off, 32);
    }
    if (lane == 0) {
        out[(size_t)n * 2 + 0] = agg3[(size_t)n * 2 + 0] + s0;
        out[(size_t)n * 2 + 1] = agg3[(size_t)n * 2 + 1] + s1;
    }
}

// ---------------------------------------------------------------------------
extern "C" void kernel_launch(void* const* d_in, const int* in_sizes, int n_in,
                              void* d_out, int out_size, void* d_ws, size_t ws_size,
                              hipStream_t stream)
{
    const float* x     = (const float*)d_in[0];
    const int*   ei    = (const int*)  d_in[1];
    const float* W1    = (const float*)d_in[2];
    const float* a1s   = (const float*)d_in[3];
    const float* a1d   = (const float*)d_in[4];
    const float* b1    = (const float*)d_in[5];
    const float* W2    = (const float*)d_in[6];
    const float* a2s   = (const float*)d_in[7];
    const float* a2d   = (const float*)d_in[8];
    const float* b2    = (const float*)d_in[9];
    const float* W3    = (const float*)d_in[10];
    const float* a3s   = (const float*)d_in[11];
    const float* a3d   = (const float*)d_in[12];
    const float* b3    = (const float*)d_in[13];
    const float* Wres1 = (const float*)d_in[14];
    const float* Wres2 = (const float*)d_in[15];
    const float* Wskip = (const float*)d_in[16];
    float* out = (float*)d_out;

    const int N  = in_sizes[0] / GAT_IN;
    const int E  = in_sizes[1] / 2;
    const int Et = E + N;

    const int Mp  = ((N + BM - 1) / BM) * BM;           // padded rows
    const int Kp1 = ((GAT_IN  + BK - 1) / BK) * BK;     // 192
    const int Kp2 = GAT_DIM;                            // 800 (already mult 32)
    const int Np  = ((GAT_DIM + BN - 1) / BN) * BN;     // 832

    // ---- workspace layout ----
    char* wsb = (char*)d_ws;
    size_t off = 0;
    auto takeF = [&](size_t n) { float* p = (float*)(wsb + off); off += n * 4; return p; };
    auto takeH = [&](size_t n) { _Float16* p = (_Float16*)(wsb + off); off += n * 2; return p; };

    float*    bufA = takeF((size_t)N * GAT_DIM);   // projections (hp)
    float*    bufB = takeF((size_t)N * GAT_DIM);   // agg1/h1 -> res+agg2/h2
    _Float16* h16  = takeH((size_t)Mp * GAT_DIM);  // padded f16 activations
    _Float16* w16  = takeH((size_t)Np * Kp2);      // padded f16 transposed W
    float*    es   = takeF((size_t)N * 8);
    float*    ed   = takeF((size_t)N * 8);
    float*    mx   = takeF((size_t)N * 8);
    float*    dn   = takeF((size_t)N * 8);
    float*    ex   = takeF((size_t)Et * 8);
    float*    h3   = takeF((size_t)N * 2);
    float*    agg3 = takeF((size_t)N * 2);
    (void)ws_size; (void)n_in; (void)out_size;

    const float NEG_INF = -INFINITY;
    const int TB = 256;
    auto cdiv = [](long long a, long long b) { return (int)((a + b - 1) / b); };

    dim3 gemmGrid(Mp / BM, Np / BN);
    const long long ND = (long long)N * GAT_DIM;

    // ================= layer 1 =================
    k_cvt_pad_f16<<<cdiv((long long)Mp * Kp1, TB), TB, 0, stream>>>(x, h16, N, GAT_IN, Mp, Kp1);
    k_cvt_w_t16  <<<cdiv((long long)Np * Kp1, TB), TB, 0, stream>>>(W1, w16, GAT_IN, GAT_DIM, Np, Kp1);
    k_gemm_wmma<<<gemmGrid, TB, 0, stream>>>(h16, w16, bufA, N, Kp1, GAT_DIM);

    k_att_coef<<<N, 256, 0, stream>>>(bufA, a1s, a1d, es, ed, N, 8, 100);
    k_fill<<<cdiv((long long)N * 8, TB), TB, 0, stream>>>(mx, (long long)N * 8, NEG_INF);
    k_fill<<<cdiv((long long)N * 8, TB), TB, 0, stream>>>(dn, (long long)N * 8, 0.0f);
    k_init_bias<<<cdiv(ND, TB), TB, 0, stream>>>(bufB, b1, ND, GAT_DIM);
    k_edge_logits<<<cdiv((long long)Et * 8, TB), TB, 0, stream>>>(ei, E, Et, es, ed, ex, mx, 8);
    k_edge_exp   <<<cdiv((long long)Et * 8, TB), TB, 0, stream>>>(ei, E, Et, ex, mx, dn, 8);
    k_edge_agg<<<cdiv(Et, 8), TB, 0, stream>>>(ei, E, Et, ex, dn, bufA, bufB, 8, 100);
    k_elu<<<cdiv(ND, TB), TB, 0, stream>>>(bufB, ND);                       // h1 in bufB

    // ================= layer 2 =================
    k_cvt_pad_f16<<<cdiv((long long)Mp * Kp2, TB), TB, 0, stream>>>(bufB, h16, N, GAT_DIM, Mp, Kp2);
    k_cvt_w_t16<<<cdiv((long long)Np * Kp2, TB), TB, 0, stream>>>(W2, w16, GAT_DIM, GAT_DIM, Np, Kp2);
    k_gemm_wmma<<<gemmGrid, TB, 0, stream>>>(h16, w16, bufA, N, Kp2, GAT_DIM);   // hp2
    k_cvt_w_t16<<<cdiv((long long)Np * Kp2, TB), TB, 0, stream>>>(Wres1, w16, GAT_DIM, GAT_DIM, Np, Kp2);
    k_gemm_wmma<<<gemmGrid, TB, 0, stream>>>(h16, w16, bufB, N, Kp2, GAT_DIM);   // res1 (over h1)
    k_add_bias<<<cdiv(ND, TB), TB, 0, stream>>>(bufB, b2, ND, GAT_DIM);          // agg2 base

    k_att_coef<<<N, 256, 0, stream>>>(bufA, a2s, a2d, es, ed, N, 8, 100);
    k_fill<<<cdiv((long long)N * 8, TB), TB, 0, stream>>>(mx, (long long)N * 8, NEG_INF);
    k_fill<<<cdiv((long long)N * 8, TB), TB, 0, stream>>>(dn, (long long)N * 8, 0.0f);
    k_edge_logits<<<cdiv((long long)Et * 8, TB), TB, 0, stream>>>(ei, E, Et, es, ed, ex, mx, 8);
    k_edge_exp   <<<cdiv((long long)Et * 8, TB), TB, 0, stream>>>(ei, E, Et, ex, mx, dn, 8);
    k_edge_agg<<<cdiv(Et, 8), TB, 0, stream>>>(ei, E, Et, ex, dn, bufA, bufB, 8, 100);
    k_elu<<<cdiv(ND, TB), TB, 0, stream>>>(bufB, ND);                       // h2 in bufB

    // ================= layer 3 =================
    k_rowdot2<<<cdiv(N, 8), TB, 0, stream>>>(bufB, W3, h3, N, GAT_DIM);     // h3 = h2@W3
    k_att_coef<<<N, 32, 0, stream>>>(h3, a3s, a3d, es, ed, N, 1, 2);
    k_fill<<<cdiv(N, TB), TB, 0, stream>>>(mx, N, NEG_INF);
    k_fill<<<cdiv(N, TB), TB, 0, stream>>>(dn, N, 0.0f);
    k_init_bias<<<cdiv((long long)N * 2, TB), TB, 0, stream>>>(agg3, b3, (long long)N * 2, 2);
    k_edge_logits<<<cdiv(Et, TB), TB, 0, stream>>>(ei, E, Et, es, ed, ex, mx, 1);
    k_edge_exp   <<<cdiv(Et, TB), TB, 0, stream>>>(ei, E, Et, ex, mx, dn, 1);
    k_edge_agg<<<cdiv(Et, 8), TB, 0, stream>>>(ei, E, Et, ex, dn, h3, agg3, 1, 2);
    k_final<<<cdiv(N, 8), TB, 0, stream>>>(agg3, bufB, Wres2, x, Wskip, out, N, GAT_DIM, GAT_IN);
}
